// MultiHeadAttention_22187801051404
// MI455X (gfx1250) — matrix operand order repaired and checked by reading, not compiled, not run on hardware
//
#include <hip/hip_runtime.h>

// ---------------------------------------------------------------------------
// Types for CDNA5 WMMA (wave32): v_wmma_f32_16x16x32_bf16
// ---------------------------------------------------------------------------
typedef __attribute__((ext_vector_type(16))) __bf16 v16bf;
typedef __attribute__((ext_vector_type(8)))  __bf16 bf16x8;
typedef __attribute__((ext_vector_type(8)))  float  v8f;

#define BB 4
#define NN 2048
#define CC 1024
#define HH 16
#define DD 64

// ---- gfx1250 async global->LDS copies (ASYNCcnt path), probe-guarded -------
#if defined(__has_builtin)
#if __has_builtin(__builtin_amdgcn_global_load_async_to_lds_b128)
#define HAVE_ASYNC_LDS 1
#endif
#endif
#ifndef HAVE_ASYNC_LDS
#define HAVE_ASYNC_LDS 0
#endif

// Builtin signature (from hipcc diagnostic): arg0 = v4i in AS1 (global),
// arg1 = v4i in AS3 (LDS), arg2 = imm offset, arg3 = imm cpol.
typedef int v4i_vs __attribute__((vector_size(16)));
typedef __attribute__((address_space(1))) v4i_vs GAS1_v4i;
typedef __attribute__((address_space(3))) v4i_vs LAS3_v4i;

// Copy 32 contiguous bytes global -> LDS without a VGPR round trip.
// Global AS1 addresses are numerically identical to generic addresses; the
// LDS byte offset is addr[31:0] of a generic shared pointer (flat aperture
// mapping), so int-casts sidestep addrspace-cast legality entirely.
__device__ __forceinline__ void async_copy_32B(const __bf16* g, __bf16* l) {
#if HAVE_ASYNC_LDS
    GAS1_v4i* gp0 = (GAS1_v4i*)(uintptr_t)g;
    GAS1_v4i* gp1 = (GAS1_v4i*)(uintptr_t)(g + 8);
    LAS3_v4i* lp0 = (LAS3_v4i*)(unsigned)(uintptr_t)l;
    LAS3_v4i* lp1 = (LAS3_v4i*)(unsigned)(uintptr_t)(l + 8);
    __builtin_amdgcn_global_load_async_to_lds_b128(gp0, lp0, 0, 0);
    __builtin_amdgcn_global_load_async_to_lds_b128(gp1, lp1, 0, 0);
#else
    (void)g; (void)l;
#endif
}

__device__ __forceinline__ void wait_async0() {
#if defined(__has_builtin) && __has_builtin(__builtin_amdgcn_s_wait_asynccnt)
    __builtin_amdgcn_s_wait_asynccnt(0);
#else
    asm volatile("s_wait_asynccnt 0x0" ::: "memory");
#endif
}

union BF16Frag { v16bf v; bf16x8 h[2]; };

// A-fragment (16x32 bf16, row-major source): lane row fixed by caller (rp points
// at [row][k0]).  ISA layout: VGPR0-3 hold K = half*8 + 0..7, VGPR4-7 hold
// K = 16 + half*8 + 0..7  -> two disjoint 16B loads.
__device__ __forceinline__ v16bf load_a_frag(const __bf16* rp, unsigned half) {
    BF16Frag f;
    f.h[0] = *reinterpret_cast<const bf16x8*>(rp + half * 8);
    f.h[1] = *reinterpret_cast<const bf16x8*>(rp + 16 + half * 8);
    return f.v;
}

// B-fragment (32x16 bf16): lane = output column; source laid out with the
// contraction dim contiguous (transposed weights / K rows / V^T rows).
// ISA layout: VGPR0-3 hold K = half*8 + 0..7, VGPR4-7 hold K = 16 + half*8 + 0..7
// -> two disjoint 16B loads (same k order as the A fragment).
__device__ __forceinline__ v16bf load_b_frag(const __bf16* rp, unsigned half) {
    BF16Frag f;
    f.h[0] = *reinterpret_cast<const bf16x8*>(rp + half * 8);
    f.h[1] = *reinterpret_cast<const bf16x8*>(rp + 16 + half * 8);
    return f.v;
}

__device__ __forceinline__ v8f wmma_bf16(v16bf a, v16bf b, v8f c) {
    return __builtin_amdgcn_wmma_f32_16x16x32_bf16(false, a, false, b,
                                                   (short)0, c, false, false);
}

// ---------------------------------------------------------------------------
// fp32 -> bf16 convert / transpose pre-passes
// ---------------------------------------------------------------------------
__global__ void cvt_f32_bf16(const float* __restrict__ in,
                             __bf16* __restrict__ out, int n) {
    int i = blockIdx.x * blockDim.x + threadIdx.x;
    if (i < n) out[i] = (__bf16)in[i];
}

// in[rows][cols] (fp32) -> out[cols][rows] (bf16)
__global__ void cvt_transpose_f32_bf16(const float* __restrict__ in,
                                       __bf16* __restrict__ out,
                                       int rows, int cols) {
    int i = blockIdx.x * blockDim.x + threadIdx.x;
    if (i < rows * cols) {
        int r = i / cols, c = i % cols;
        out[(size_t)c * rows + r] = (__bf16)in[i];
    }
}

// ---------------------------------------------------------------------------
// GEMM mainloop with LDS-staged, double-buffered B tile.
// Block = 256 threads = 8 waves arranged 4(rows) x 2(cols); block tile
// 128(M) x 128(N); k-step 32.  Wave tile = 32x64: 2 M-reps x 4 N-tiles, so
// each B fragment feeds TWO back-to-back WMMAs and each A fragment feeds four
// (8 WMMAs per 8 ds_load_b128 + 4 global_load_b128 per k-step).
// B staging uses GLOBAL_LOAD_ASYNC_TO_LDS_B128 when available (ASYNCcnt,
// no VGPR round trip), falling back to global_load + ds_store.  The next
// tile's copy is issued at the top of the iteration: the previous barrier
// guarantees all waves finished reading that buffer.
// ---------------------------------------------------------------------------
#define LDS_B_ELEMS (128 * 32)   // one buffer: 128 cols x 32 k-elems (8KB)

__device__ __forceinline__ void gemm_wave_lds(const __bf16* __restrict__ A, int lda,
                                              const __bf16* __restrict__ Bt, int ldb,
                                              int rowBaseWave, int colBaseBlock,
                                              int waveColOff, int Kdim,
                                              v8f acc[2][4], __bf16* ldsB) {
    const int tid  = threadIdx.x;
    const int lane = tid & 31;
    const int m    = lane & 15;
    const unsigned half = (unsigned)(lane >> 4);

    // Cooperative staging: thread -> (col = tid/2, 16-elem half of the k-row).
    const int ldcol = tid >> 1;
    const int ldoff = (tid & 1) * 16;
    const __bf16* gsrc = Bt + (size_t)(colBaseBlock + ldcol) * ldb + ldoff;
    __bf16* ldst = ldsB + ldcol * 32 + ldoff;

    // Stage k-tile 0 into buffer 0.
#if HAVE_ASYNC_LDS
    async_copy_32B(gsrc, ldst);
    wait_async0();
#else
    {
        bf16x8 s0 = *reinterpret_cast<const bf16x8*>(gsrc);
        bf16x8 s1 = *reinterpret_cast<const bf16x8*>(gsrc + 8);
        *reinterpret_cast<bf16x8*>(ldst)     = s0;
        *reinterpret_cast<bf16x8*>(ldst + 8) = s1;
    }
#endif
    __syncthreads();

    const __bf16* arow0 = A + (size_t)(rowBaseWave + m) * lda;
    const __bf16* arow1 = A + (size_t)(rowBaseWave + 16 + m) * lda;
    int buf = 0;
    for (int k0 = 0; k0 < Kdim; k0 += 32) {
        const bool haveNext = (k0 + 32 < Kdim);
#if HAVE_ASYNC_LDS
        // Issue next tile's async global->LDS copy; lands while WMMAs run.
        if (haveNext)
            async_copy_32B(gsrc + k0 + 32, ldst + (buf ^ 1) * LDS_B_ELEMS);
#else
        bf16x8 n0, n1;
        if (haveNext) {
            n0 = *reinterpret_cast<const bf16x8*>(gsrc + k0 + 32);
            n1 = *reinterpret_cast<const bf16x8*>(gsrc + k0 + 40);
        }
#endif
        // A fragments (2 M-reps) from global; B fragments (4 N-tiles) from LDS.
        const v16bf a0 = load_a_frag(arow0 + k0, half);
        const v16bf a1 = load_a_frag(arow1 + k0, half);
        const __bf16* lb = ldsB + buf * LDS_B_ELEMS + (waveColOff + m) * 32;
        v16bf b0 = load_b_frag(lb + 0 * 16 * 32, half);
        v16bf b1 = load_b_frag(lb + 1 * 16 * 32, half);
        v16bf b2 = load_b_frag(lb + 2 * 16 * 32, half);
        v16bf b3 = load_b_frag(lb + 3 * 16 * 32, half);
        // 8 independent WMMAs; each B fragment used twice back-to-back.
        acc[0][0] = wmma_bf16(a0, b0, acc[0][0]);
        acc[1][0] = wmma_bf16(a1, b0, acc[1][0]);
        acc[0][1] = wmma_bf16(a0, b1, acc[0][1]);
        acc[1][1] = wmma_bf16(a1, b1, acc[1][1]);
        acc[0][2] = wmma_bf16(a0, b2, acc[0][2]);
        acc[1][2] = wmma_bf16(a1, b2, acc[1][2]);
        acc[0][3] = wmma_bf16(a0, b3, acc[0][3]);
        acc[1][3] = wmma_bf16(a1, b3, acc[1][3]);
        if (haveNext) {
#if HAVE_ASYNC_LDS
            wait_async0();     // this wave's async writes have landed in LDS
#else
            __bf16* dst = ldst + (buf ^ 1) * LDS_B_ELEMS;
            *reinterpret_cast<bf16x8*>(dst)     = n0;
            *reinterpret_cast<bf16x8*>(dst + 8) = n1;
#endif
            __syncthreads();   // orders iter-i reads(buf) vs iter-i+1 stores(buf)
            buf ^= 1;
        }
    }
}

// ---------------------------------------------------------------------------
// QKV projection: X[8192,1024]bf16 @ Wqkv^T[3072,1024]bf16 + b.
// Epilogue scatters Q,K -> [B,H,N,D] and V -> [B,H,D,N] (transposed) in bf16.
// ---------------------------------------------------------------------------
__global__ void __launch_bounds__(256, 1)
qkv_gemm_kernel(const __bf16* __restrict__ X, const __bf16* __restrict__ WqkvT,
                const float* __restrict__ bqkv,
                __bf16* __restrict__ Qb, __bf16* __restrict__ Kb,
                __bf16* __restrict__ Vtb) {
    __shared__ __align__(16) __bf16 ldsB[2 * LDS_B_ELEMS];   // 16KB
    v8f acc[2][4] = {};
    const int wave = threadIdx.x >> 5;
    const int lane = threadIdx.x & 31;
    const int waveRow = wave & 3;
    const int waveColOff = (wave >> 2) * 64;
    const int rowBaseWave = blockIdx.y * 128 + waveRow * 32;
    const int colBaseBlock = blockIdx.x * 128;
    gemm_wave_lds(X, CC, WqkvT, CC, rowBaseWave, colBaseBlock, waveColOff,
                  CC, acc, ldsB);

    const int ncol = lane & 15;
    const unsigned half = (unsigned)(lane >> 4);
#pragma unroll
    for (int mr = 0; mr < 2; ++mr) {
#pragma unroll
        for (int t = 0; t < 4; ++t) {
            const int c = colBaseBlock + waveColOff + t * 16 + ncol;
            const float bias = bqkv[c];
            const int s = c >> 10;           // 0:q 1:k 2:v
            const int cc = c & (CC - 1);
            const int h = cc >> 6;
            const int d = cc & (DD - 1);
#pragma unroll
            for (int r = 0; r < 8; ++r) {
                const int g = rowBaseWave + mr * 16 + r + 8 * (int)half;
                const int b = g >> 11;
                const int n = g & (NN - 1);
                const __bf16 bv = (__bf16)(acc[mr][t][r] + bias);
                const size_t bh = (size_t)(b * HH + h);
                if (s == 0)      Qb[(bh * NN + n) * DD + d] = bv;
                else if (s == 1) Kb[(bh * NN + n) * DD + d] = bv;
                else             Vtb[(bh * DD + d) * NN + n] = bv;
            }
        }
    }
}

// ---------------------------------------------------------------------------
// Flash attention: one wave = 16 queries of one (b,h); stream 32 keys / iter.
// S = Q K^T (4 WMMAs), online softmax (shfl_xor row reductions + v_exp_f32),
// P through wave-private LDS (layout conversion), O += P V (4 WMMAs).
// V fragments are fetched before the softmax block so their global-load
// latency hides under the exp/shuffle VALU work.  No barriers needed: each
// wave touches only its own LDS region and per-wave LDS ops are in-order.
// ---------------------------------------------------------------------------
__global__ void __launch_bounds__(256, 1)
flash_attn_kernel(const __bf16* __restrict__ Qb, const __bf16* __restrict__ Kb,
                  const __bf16* __restrict__ Vtb, __bf16* __restrict__ attnOut) {
    __shared__ __align__(16) __bf16 lds_p[8][16 * 32];   // 8KB: per-wave P tile

    const int wave = threadIdx.x >> 5;
    const int lane = threadIdx.x & 31;
    const int bh = blockIdx.x;           // 0..63 == b*H + h
    const int b = bh >> 4;
    const int h = bh & (HH - 1);
    const int qBase = blockIdx.y * 128 + wave * 16;

    const __bf16* Q  = Qb  + (size_t)bh * NN * DD;
    const __bf16* K  = Kb  + (size_t)bh * NN * DD;
    const __bf16* Vt = Vtb + (size_t)bh * DD * NN;

    const int m = lane & 15;
    const unsigned half = (unsigned)(lane >> 4);

    const __bf16* qrow = Q + (size_t)(qBase + m) * DD;
    const v16bf qf0 = load_a_frag(qrow + 0,  half);
    const v16bf qf1 = load_a_frag(qrow + 32, half);

    v8f o[4] = {};
    float mi[8], li[8];
#pragma unroll
    for (int r = 0; r < 8; ++r) { mi[r] = -3.0e38f; li[r] = 0.0f; }

    const float scale = 0.125f;          // 1/sqrt(64)
    const float L2E = 1.44269504f;

    for (int kt = 0; kt < NN; kt += 32) {
        // ---- preload K fragments, then S = Q K^T (two 16x16 tiles) ----
        const __bf16* kr0 = K + (size_t)(kt + m) * DD;
        const __bf16* kr1 = K + (size_t)(kt + 16 + m) * DD;
        v16bf kb[4];
        kb[0] = load_b_frag(kr0 + 0,  half);
        kb[1] = load_b_frag(kr0 + 32, half);
        kb[2] = load_b_frag(kr1 + 0,  half);
        kb[3] = load_b_frag(kr1 + 32, half);
        v8f s0 = {}, s1 = {};
        s0 = wmma_bf16(qf0, kb[0], s0);
        s0 = wmma_bf16(qf1, kb[1], s0);
        s1 = wmma_bf16(qf0, kb[2], s1);
        s1 = wmma_bf16(qf1, kb[3], s1);

        // ---- prefetch V fragments: overlap global loads with softmax ----
        v16bf vb[4];
#pragma unroll
        for (int t = 0; t < 4; ++t) {
            const __bf16* vrow = Vt + (size_t)(t * 16 + m) * NN + kt;
            vb[t] = load_b_frag(vrow, half);
        }

        // ---- online softmax ----
        float alpha[8];
#pragma unroll
        for (int r = 0; r < 8; ++r) {
            const float v0 = s0[r] * scale;
            const float v1 = s1[r] * scale;
            float rowmax = fmaxf(v0, v1);
#pragma unroll
            for (int off = 1; off < 16; off <<= 1)
                rowmax = fmaxf(rowmax, __shfl_xor(rowmax, off, 32));
            const float mnew = fmaxf(mi[r], rowmax);
            alpha[r] = __builtin_amdgcn_exp2f(L2E * (mi[r] - mnew));
            const float p0 = __builtin_amdgcn_exp2f(L2E * (v0 - mnew));
            const float p1 = __builtin_amdgcn_exp2f(L2E * (v1 - mnew));
            float rs = p0 + p1;
#pragma unroll
            for (int off = 1; off < 16; off <<= 1)
                rs += __shfl_xor(rs, off, 32);
            li[r] = li[r] * alpha[r] + rs;
            mi[r] = mnew;
            // C-layout element -> LDS row-major P tile (wave-private region)
            const int row = r + 8 * (int)half;
            lds_p[wave][row * 32 + (lane & 15)]      = (__bf16)p0;
            lds_p[wave][row * 32 + 16 + (lane & 15)] = (__bf16)p1;
        }
#pragma unroll
        for (int t = 0; t < 4; ++t)
#pragma unroll
            for (int r = 0; r < 8; ++r) o[t][r] *= alpha[r];

        // ---- O += P V (P re-read as A-fragment; per-wave LDS is in-order) --
        const v16bf pf = load_a_frag(&lds_p[wave][m * 32], half);
#pragma unroll
        for (int t = 0; t < 4; ++t)
            o[t] = wmma_bf16(pf, vb[t], o[t]);
    }

    // ---- normalize + store bf16 into [B*N, C] ----
    float inv[8];
#pragma unroll
    for (int r = 0; r < 8; ++r) inv[r] = 1.0f / li[r];
#pragma unroll
    for (int t = 0; t < 4; ++t) {
#pragma unroll
        for (int r = 0; r < 8; ++r) {
            const int row = r + 8 * (int)half;
            const int q = qBase + row;
            const int dcol = t * 16 + (lane & 15);
            attnOut[((size_t)(b * NN + q)) * CC + h * DD + dcol] =
                (__bf16)(o[t][r] * inv[r]);
        }
    }
}

// ---------------------------------------------------------------------------
// Output projection: attn[8192,1024]bf16 @ Wproj^T[1024,1024]bf16 + b -> fp32
// ---------------------------------------------------------------------------
__global__ void __launch_bounds__(256, 1)
proj_gemm_kernel(const __bf16* __restrict__ Attn, const __bf16* __restrict__ WprojT,
                 const float* __restrict__ bproj, float* __restrict__ out) {
    __shared__ __align__(16) __bf16 ldsB[2 * LDS_B_ELEMS];   // 16KB
    v8f acc[2][4] = {};
    const int wave = threadIdx.x >> 5;
    const int lane = threadIdx.x & 31;
    const int waveRow = wave & 3;
    const int waveColOff = (wave >> 2) * 64;
    const int rowBaseWave = blockIdx.y * 128 + waveRow * 32;
    const int colBaseBlock = blockIdx.x * 128;
    gemm_wave_lds(Attn, CC, WprojT, CC, rowBaseWave, colBaseBlock, waveColOff,
                  CC, acc, ldsB);

    const int ncol = lane & 15;
    const unsigned half = (unsigned)(lane >> 4);
#pragma unroll
    for (int mr = 0; mr < 2; ++mr) {
#pragma unroll
        for (int t = 0; t < 4; ++t) {
            const int c = colBaseBlock + waveColOff + t * 16 + ncol;
            const float bias = bproj[c];
#pragma unroll
            for (int r = 0; r < 8; ++r) {
                const int g = rowBaseWave + mr * 16 + r + 8 * (int)half;
                out[(size_t)g * CC + c] = acc[mr][t][r] + bias;
            }
        }
    }
}

// ---------------------------------------------------------------------------
// Host launcher
// ---------------------------------------------------------------------------
extern "C" void kernel_launch(void* const* d_in, const int* in_sizes, int n_in,
                              void* d_out, int out_size, void* d_ws, size_t ws_size,
                              hipStream_t stream) {
    const float* x_q    = (const float*)d_in[0];   // [B,N,C]
    const float* w_qkv  = (const float*)d_in[1];   // [C,3C]
    const float* b_qkv  = (const float*)d_in[2];   // [3C]
    const float* w_proj = (const float*)d_in[3];   // [C,C]
    const float* b_proj = (const float*)d_in[4];   // [C]
    float* out = (float*)d_out;                    // [B,N,C] fp32

    const size_t M = (size_t)BB * NN;              // 8192
    char* ws = (char*)d_ws;
    size_t off = 0;
    __bf16* Xbf    = (__bf16*)(ws + off); off += M * CC * 2;               // 16MB
    __bf16* WqkvT  = (__bf16*)(ws + off); off += (size_t)3 * CC * CC * 2;  // 6MB
    __bf16* WprojT = (__bf16*)(ws + off); off += (size_t)CC * CC * 2;      // 2MB
    __bf16* Qb     = (__bf16*)(ws + off); off += M * CC * 2;               // 16MB
    __bf16* Kb     = (__bf16*)(ws + off); off += M * CC * 2;               // 16MB
    __bf16* Vtb    = (__bf16*)(ws + off); off += M * CC * 2;               // 16MB
    __bf16* Attn   = (__bf16*)(ws + off); off += M * CC * 2;               // 16MB
    (void)ws_size; (void)in_sizes; (void)n_in; (void)out_size;

    // 1) precision / layout conversion
    {
        int n = (int)(M * CC);
        cvt_f32_bf16<<<(n + 255) / 256, 256, 0, stream>>>(x_q, Xbf, n);
        cvt_transpose_f32_bf16<<<(CC * 3 * CC + 255) / 256, 256, 0, stream>>>(
            w_qkv, WqkvT, CC, 3 * CC);
        cvt_transpose_f32_bf16<<<(CC * CC + 255) / 256, 256, 0, stream>>>(
            w_proj, WprojT, CC, CC);
    }
    // 2) fused QKV projection with Q/K/V layout epilogue
    {
        dim3 grid(3 * CC / 128, (unsigned)(M / 128));   // 24 x 64
        qkv_gemm_kernel<<<grid, 256, 0, stream>>>(Xbf, WqkvT, b_qkv, Qb, Kb, Vtb);
    }
    // 3) flash attention
    {
        dim3 grid(BB * HH, NN / 128);                   // 64 x 16
        flash_attn_kernel<<<grid, 256, 0, stream>>>(Qb, Kb, Vtb, Attn);
    }
    // 4) output projection (fp32 out + bias)
    {
        dim3 grid(CC / 128, (unsigned)(M / 128));       // 8 x 64
        proj_gemm_kernel<<<grid, 256, 0, stream>>>(Attn, WprojT, b_proj, out);
    }
}
